// GAT_39324720562942
// MI455X (gfx1250) — matrix operand order repaired and checked by reading, not compiled, not run on hardware
//
#include <hip/hip_runtime.h>
#include <hip/hip_bf16.h>
#include <math.h>

typedef __bf16 bf16;
typedef bf16  v16bf __attribute__((ext_vector_type(16)));
typedef float v8f   __attribute__((ext_vector_type(8)));

union Frag { v16bf v; float4 q[2]; };

#define HEADS 4
#define CH    256   // per-head channels (all conv layers)
#define NEG_SLOPE 0.2f

// GEMM wave tile: 64 rows x 32 cols (MT=4, NT=2) -> 8 WMMAs per K-step
#define MT 4
#define NT 2
#define TILE_M (MT * 16)
#define TILE_N (NT * 16)

// ---------------------------------------------------------------------------
// utility kernels
// ---------------------------------------------------------------------------
__global__ void k_cvt_bf16(const float* __restrict__ s, bf16* __restrict__ d, size_t n) {
    size_t i = (size_t)blockIdx.x * blockDim.x + threadIdx.x;
    if (i < n) d[i] = (bf16)s[i];
}

__global__ void k_fill(float* __restrict__ p, float v, size_t n) {
    size_t i = (size_t)blockIdx.x * blockDim.x + threadIdx.x;
    if (i < n) p[i] = v;
}

// |x - q| -> bf16, x:[N,1024], q:[1024]
__global__ void k_xdist(const float* __restrict__ x, const float* __restrict__ q,
                        bf16* __restrict__ xd, size_t total) {
    size_t t = (size_t)blockIdx.x * blockDim.x + threadIdx.x;
    if (t >= total) return;
    int c = (int)(t & 1023);
    xd[t] = (bf16)fabsf(x[t] - q[c]);
}

// x * xw -> bf16, x:[N,1024], xw:[N]
__global__ void k_gate(const float* __restrict__ x, const float* __restrict__ xw,
                       bf16* __restrict__ xg, size_t total) {
    size_t t = (size_t)blockIdx.x * blockDim.x + threadIdx.x;
    if (t >= total) return;
    size_t n = t >> 10;
    xg[t] = (bf16)(x[t] * xw[n]);
}

// ---------------------------------------------------------------------------
// WMMA bf16 GEMM:  C[M,Ncols] = act( A[M,K] @ W[Ncols,K]^T + bias )
// one wave per 64x32 output tile; K multiple of 32; Ncols multiple of 32
//
// Fragment layouts (ISA 7.12.2, wave32):
//   A 16x32: lane l -> row l&15; elem e -> K = (e&7) + 8*(l>>4) + 16*(e>>3)
//            => two contiguous 16B loads at kb+8*half and kb+8*half+16
//   B 32x16: lane l -> col l&15; elem e -> K = (e&7) + 8*(l>>4) + 16*(e>>3)
//            => two contiguous 16B loads at kb+8*half and kb+8*half+16
//   C/D    : lane l -> col l&15; VGPR r -> row r + 8*(l>>4)
// ---------------------------------------------------------------------------
__device__ __forceinline__ v16bf load_fragA(const bf16* __restrict__ base, int stride,
                                            int row0, int rmax, int kb, int lane) {
    int m = lane & 15, half = lane >> 4;
    int r = row0 + m; if (r > rmax) r = rmax;
    const bf16* p = base + (size_t)r * stride + kb + half * 8;
    Frag f;
    f.q[0] = *(const float4*)(p);
    f.q[1] = *(const float4*)(p + 16);
    return f.v;
}

__device__ __forceinline__ v16bf load_fragB(const bf16* __restrict__ base, int stride,
                                            int col0, int kb, int lane) {
    int n = lane & 15, half = lane >> 4;
    const bf16* p = base + (size_t)(col0 + n) * stride + kb + half * 8;
    Frag f;
    f.q[0] = *(const float4*)(p);
    f.q[1] = *(const float4*)(p + 16);
    return f.v;
}

__global__ __launch_bounds__(32)
void k_gemm_bf16(const bf16* __restrict__ A, const bf16* __restrict__ W,
                 const float* __restrict__ bias,
                 float* __restrict__ Cf, bf16* __restrict__ Cb,
                 int M, int Ncols, int K, int relu) {
    int lane = threadIdx.x;
    int row0 = blockIdx.x * TILE_M;
    int col0 = blockIdx.y * TILE_N;
    v8f acc[MT][NT] = {};
    for (int kb = 0; kb < K; kb += 32) {
        v16bf a[MT], b[NT];
        #pragma unroll
        for (int mt = 0; mt < MT; ++mt)
            a[mt] = load_fragA(A, K, row0 + 16 * mt, M - 1, kb, lane);
        #pragma unroll
        for (int nt = 0; nt < NT; ++nt)
            b[nt] = load_fragB(W, K, col0 + 16 * nt, kb, lane);
        #pragma unroll
        for (int mt = 0; mt < MT; ++mt)
            #pragma unroll
            for (int nt = 0; nt < NT; ++nt)
                acc[mt][nt] = __builtin_amdgcn_wmma_f32_16x16x32_bf16(
                    false, a[mt], false, b[nt], (short)0, acc[mt][nt], false, false);
    }
    int n_lo  = lane & 15;
    int rbase = (lane >> 4) * 8;
    #pragma unroll
    for (int mt = 0; mt < MT; ++mt) {
        #pragma unroll
        for (int nt = 0; nt < NT; ++nt) {
            int col = col0 + nt * 16 + n_lo;
            float bv = bias ? bias[col] : 0.0f;
            #pragma unroll
            for (int r = 0; r < 8; ++r) {
                int row = row0 + mt * 16 + rbase + r;
                if (row < M) {
                    float v = acc[mt][nt][r] + bv;
                    if (relu) v = fmaxf(v, 0.0f);
                    size_t off = (size_t)row * Ncols + col;
                    if (Cf) Cf[off] = v;
                    if (Cb) Cb[off] = (bf16)v;
                }
            }
        }
    }
}

// ---------------------------------------------------------------------------
// MLP final layer: xw[n] = sigmoid( act2[n,:512] . W3 + b3 )
// ---------------------------------------------------------------------------
__global__ void k_mlp3(const float* __restrict__ a2, const float* __restrict__ w3,
                       const float* __restrict__ b3, float* __restrict__ xw, int N) {
    int n = blockIdx.x * blockDim.x + threadIdx.x;
    if (n >= N) return;
    const float* a = a2 + (size_t)n * 512;
    float s = 0.0f;
    for (int c = 0; c < 512; ++c) s += a[c] * w3[c];
    s += b3[0];
    xw[n] = 1.0f / (1.0f + __expf(-s));
}

// ---------------------------------------------------------------------------
// attention per-node scores: a_src/a_dst [N,4]
// ---------------------------------------------------------------------------
__global__ void k_att(const float* __restrict__ h, const float* __restrict__ att_s,
                      const float* __restrict__ att_d, float* __restrict__ a_s,
                      float* __restrict__ a_d, int N) {
    int t = blockIdx.x * blockDim.x + threadIdx.x;
    if (t >= N * HEADS) return;
    int n = t >> 2, hd = t & 3;
    const float* hp = h + (size_t)n * (HEADS * CH) + hd * CH;
    const float* ws = att_s + hd * CH;
    const float* wd = att_d + hd * CH;
    float s = 0.0f, d = 0.0f;
    for (int c = 0; c < CH; ++c) { float v = hp[c]; s += v * ws[c]; d += v * wd[c]; }
    a_s[t] = s; a_d[t] = d;
}

// ---------------------------------------------------------------------------
// edge kernels  (Etot = E + N, self loops appended)
// ---------------------------------------------------------------------------
__device__ __forceinline__ void atomicMaxFloat(float* addr, float val) {
    if (val >= 0.0f) atomicMax((int*)addr, __float_as_int(val));
    else             atomicMin((unsigned int*)addr, (unsigned int)__float_as_int(val));
}

__global__ void k_edge_logits(const int* __restrict__ ei, int E, int N,
                              const float* __restrict__ a_s, const float* __restrict__ a_d,
                              float* __restrict__ ebuf, float* __restrict__ emax) {
    int i = blockIdx.x * blockDim.x + threadIdx.x;
    if (i >= E + N) return;
    int s, d;
    if (i < E) { s = ei[i]; d = ei[E + i]; } else { s = d = i - E; }
    #pragma unroll
    for (int hd = 0; hd < HEADS; ++hd) {
        float e = a_s[s * HEADS + hd] + a_d[d * HEADS + hd];
        e = (e > 0.0f) ? e : NEG_SLOPE * e;
        ebuf[(size_t)i * HEADS + hd] = e;
        atomicMaxFloat(&emax[d * HEADS + hd], e);
    }
}

__global__ void k_edge_exp(const int* __restrict__ ei, int E, int N,
                           float* __restrict__ ebuf, const float* __restrict__ emax,
                           float* __restrict__ denom) {
    int i = blockIdx.x * blockDim.x + threadIdx.x;
    if (i >= E + N) return;
    int d;
    if (i < E) { d = ei[E + i]; } else { d = i - E; }
    #pragma unroll
    for (int hd = 0; hd < HEADS; ++hd) {
        float ee = __expf(ebuf[(size_t)i * HEADS + hd] - emax[d * HEADS + hd]);
        ebuf[(size_t)i * HEADS + hd] = ee;
        atomicAdd(&denom[d * HEADS + hd], ee);
    }
}

// one block (256 threads) per edge; scatter-add messages into agg[N,4,256]
__global__ __launch_bounds__(256)
void k_edge_msg(const int* __restrict__ ei, int E, int N,
                const float* __restrict__ h, const float* __restrict__ ebuf,
                const float* __restrict__ denom, float* __restrict__ agg) {
    int i = blockIdx.x;
    int c = threadIdx.x;
    int s, d;
    if (i < E) { s = ei[i]; d = ei[E + i]; } else { s = d = i - E; }
    #pragma unroll
    for (int hd = 0; hd < HEADS; ++hd) {
        float alpha = ebuf[(size_t)i * HEADS + hd] / (denom[d * HEADS + hd] + 1e-16f);
        atomicAdd(&agg[(size_t)d * (HEADS * CH) + hd * CH + c],
                  h[(size_t)s * (HEADS * CH) + hd * CH + c] * alpha);
    }
}

// head mean + bias: out[n,c] = mean_h agg[n,h,c] + bias[c]
__global__ void k_mean_bias(const float* __restrict__ agg, const float* __restrict__ bias,
                            float* __restrict__ out, size_t total) {
    size_t t = (size_t)blockIdx.x * blockDim.x + threadIdx.x;
    if (t >= total) return;
    size_t n = t >> 8;
    int c = (int)(t & 255);
    const float* a = agg + n * (HEADS * CH);
    float v = 0.25f * (a[c] + a[CH + c] + a[2 * CH + c] + a[3 * CH + c]) + bias[c];
    out[t] = v;
}

// ---------------------------------------------------------------------------
// batchnorm (train-style, biased var) + relu -> bf16
// ---------------------------------------------------------------------------
__global__ __launch_bounds__(256)
void k_bn_stats(const float* __restrict__ xin, float* __restrict__ bsum,
                float* __restrict__ bsq, int N) {
    int c = threadIdx.x;
    float s = 0.0f, q = 0.0f;
    for (int n = blockIdx.x; n < N; n += gridDim.x) {
        float v = xin[(size_t)n * CH + c];
        s += v; q += v * v;
    }
    atomicAdd(&bsum[c], s);
    atomicAdd(&bsq[c], q);
}

__global__ void k_bn_apply(const float* __restrict__ xin, const float* __restrict__ bsum,
                           const float* __restrict__ bsq, const float* __restrict__ gamma,
                           const float* __restrict__ beta, bf16* __restrict__ xout,
                           int N, size_t total) {
    size_t t = (size_t)blockIdx.x * blockDim.x + threadIdx.x;
    if (t >= total) return;
    int c = (int)(t & 255);
    float invN = 1.0f / (float)N;
    float mu  = bsum[c] * invN;
    float var = bsq[c] * invN - mu * mu;
    float v = (xin[t] - mu) * rsqrtf(var + 1e-5f) * gamma[c] + beta[c];
    v = fmaxf(v, 0.0f);
    xout[t] = (bf16)v;
}

// ---------------------------------------------------------------------------
// host-side orchestration
// ---------------------------------------------------------------------------
static inline char* carve(char*& p, size_t bytes) {
    char* r = p;
    p += (bytes + 255) & ~(size_t)255;
    return r;
}
static inline dim3 g1(size_t n, int b) { return dim3((unsigned)((n + b - 1) / b)); }

extern "C" void kernel_launch(void* const* d_in, const int* in_sizes, int n_in,
                              void* d_out, int out_size, void* d_ws, size_t ws_size,
                              hipStream_t stream) {
    const float* x         = (const float*)d_in[0];
    const int*   ei        = (const int*)d_in[1];
    const float* qnode     = (const float*)d_in[2];
    const float* edge_attr = (const float*)d_in[3];
    // d_in[4] = question_edge (unused: edge_attr all zero -> edge gating skipped)
    const float* convW[3]  = { (const float*)d_in[5], (const float*)d_in[9],  (const float*)d_in[13] };
    const float* convAS[3] = { (const float*)d_in[6], (const float*)d_in[10], (const float*)d_in[14] };
    const float* convAD[3] = { (const float*)d_in[7], (const float*)d_in[11], (const float*)d_in[15] };
    const float* convB[3]  = { (const float*)d_in[8], (const float*)d_in[12], (const float*)d_in[16] };
    const float* bnG[2]    = { (const float*)d_in[17], (const float*)d_in[19] };
    const float* bnB[2]    = { (const float*)d_in[18], (const float*)d_in[20] };
    const float* mW1 = (const float*)d_in[21]; const float* mb1 = (const float*)d_in[22];
    const float* mW2 = (const float*)d_in[23]; const float* mb2 = (const float*)d_in[24];
    const float* mW3 = (const float*)d_in[25]; const float* mb3 = (const float*)d_in[26];

    const int N    = in_sizes[0] / 1024;   // 20000
    const int E    = in_sizes[1] / 2;      // 320000
    const int Etot = E + N;
    const int IN_C = 1024, HC = HEADS * CH; // 1024

    // ---- workspace carve (with lifetime aliasing) ----
    char* p = (char*)d_ws;
    float* h_f32   = (float*)carve(p, (size_t)N * HC * 4);    // also aliases xd (bf16 N*1024 fits)
    float* agg     = (float*)carve(p, (size_t)N * HC * 4);    // also aliases act2 (f32 N*512 fits)
    bf16*  xg      = (bf16*) carve(p, (size_t)N * IN_C * 2);
    bf16*  act1    = (bf16*) carve(p, (size_t)N * 512 * 2);
    float* xw      = (float*)carve(p, (size_t)N * 4);
    float* a_src   = (float*)carve(p, (size_t)N * HEADS * 4);
    float* a_dst   = (float*)carve(p, (size_t)N * HEADS * 4);
    float* ebuf    = (float*)carve(p, (size_t)Etot * HEADS * 4);
    float* emax    = (float*)carve(p, (size_t)N * HEADS * 4);
    float* denom   = (float*)carve(p, (size_t)N * HEADS * 4);
    float* xcur_f  = (float*)carve(p, (size_t)N * CH * 4);
    bf16*  xcur_b  = (bf16*) carve(p, (size_t)N * CH * 2);
    float* bnsum   = (float*)carve(p, 256 * 4);
    float* bnsq    = (float*)carve(p, 256 * 4);
    bf16*  wm1b    = (bf16*) carve(p, (size_t)512 * 1024 * 2);
    bf16*  wm2b    = (bf16*) carve(p, (size_t)512 * 512 * 2);
    bf16*  wc_b[3];
    wc_b[0] = (bf16*)carve(p, (size_t)1024 * 1024 * 2);
    wc_b[1] = (bf16*)carve(p, (size_t)1024 * 256 * 2);
    wc_b[2] = (bf16*)carve(p, (size_t)1024 * 256 * 2);
    bf16*  xd   = (bf16*) h_f32;   // alias: dead before h is written
    float* act2 = (float*)agg;     // alias: dead before agg is used

    // ---- weight conversion f32 -> bf16 ----
    k_cvt_bf16<<<g1((size_t)512 * 1024, 256), 256, 0, stream>>>(mW1, wm1b, (size_t)512 * 1024);
    k_cvt_bf16<<<g1((size_t)512 * 512, 256), 256, 0, stream>>>(mW2, wm2b, (size_t)512 * 512);
    k_cvt_bf16<<<g1((size_t)1024 * 1024, 256), 256, 0, stream>>>(convW[0], wc_b[0], (size_t)1024 * 1024);
    k_cvt_bf16<<<g1((size_t)1024 * 256, 256), 256, 0, stream>>>(convW[1], wc_b[1], (size_t)1024 * 256);
    k_cvt_bf16<<<g1((size_t)1024 * 256, 256), 256, 0, stream>>>(convW[2], wc_b[2], (size_t)1024 * 256);

    const int rowTiles = (N + TILE_M - 1) / TILE_M;

    // ---- gating MLP ----
    k_xdist<<<g1((size_t)N * IN_C, 256), 256, 0, stream>>>(x, qnode, xd, (size_t)N * IN_C);
    k_gemm_bf16<<<dim3(rowTiles, 512 / TILE_N), 32, 0, stream>>>(xd, wm1b, mb1, nullptr, act1, N, 512, 1024, 1);
    k_gemm_bf16<<<dim3(rowTiles, 512 / TILE_N), 32, 0, stream>>>(act1, wm2b, mb2, act2, nullptr, N, 512, 512, 1);
    k_mlp3<<<g1((size_t)N, 256), 256, 0, stream>>>(act2, mW3, mb3, xw, N);
    k_gate<<<g1((size_t)N * IN_C, 256), 256, 0, stream>>>(x, xw, xg, (size_t)N * IN_C);

    // ---- three GAT conv layers ----
    for (int l = 0; l < 3; ++l) {
        const bf16* Ain = (l == 0) ? xg : xcur_b;
        const int K = (l == 0) ? 1024 : 256;
        k_gemm_bf16<<<dim3(rowTiles, HC / TILE_N), 32, 0, stream>>>(Ain, wc_b[l], nullptr, h_f32, nullptr, N, HC, K, 0);
        k_att<<<g1((size_t)N * HEADS, 256), 256, 0, stream>>>(h_f32, convAS[l], convAD[l], a_src, a_dst, N);

        k_fill<<<g1((size_t)N * HEADS, 256), 256, 0, stream>>>(emax, -1e30f, (size_t)N * HEADS);
        k_fill<<<g1((size_t)N * HEADS, 256), 256, 0, stream>>>(denom, 0.0f, (size_t)N * HEADS);
        k_fill<<<g1((size_t)N * HC, 256), 256, 0, stream>>>(agg, 0.0f, (size_t)N * HC);

        k_edge_logits<<<g1((size_t)Etot, 256), 256, 0, stream>>>(ei, E, N, a_src, a_dst, ebuf, emax);
        k_edge_exp<<<g1((size_t)Etot, 256), 256, 0, stream>>>(ei, E, N, ebuf, emax, denom);
        k_edge_msg<<<dim3(Etot), 256, 0, stream>>>(ei, E, N, h_f32, ebuf, denom, agg);

        float* mean_out = (l == 2) ? (float*)d_out : xcur_f;
        k_mean_bias<<<g1((size_t)N * CH, 256), 256, 0, stream>>>(agg, convB[l], mean_out, (size_t)N * CH);

        if (l < 2) {
            k_fill<<<1, 256, 0, stream>>>(bnsum, 0.0f, 256);
            k_fill<<<1, 256, 0, stream>>>(bnsq, 0.0f, 256);
            k_bn_stats<<<dim3(128), 256, 0, stream>>>(xcur_f, bnsum, bnsq, N);
            k_bn_apply<<<g1((size_t)N * CH, 256), 256, 0, stream>>>(xcur_f, bnsum, bnsq, bnG[l], bnB[l],
                                                                    xcur_b, N, (size_t)N * CH);
        }
    }

    // ---- second output: edge_attr passthrough (all-zero) ----
    hipMemcpyAsync((float*)d_out + (size_t)N * CH, edge_attr,
                   (size_t)E * IN_C * sizeof(float), hipMemcpyDeviceToDevice, stream);
}
